// GENGCNNetwork_68186900791434
// MI455X (gfx1250) — matrix operand order repaired and checked by reading, not compiled, not run on hardware
//
#include <hip/hip_runtime.h>
#include <hip/hip_bf16.h>
#include <math.h>
#include <stdint.h>

typedef __attribute__((ext_vector_type(16))) __bf16 v16bf;
typedef __attribute__((ext_vector_type(8)))  __bf16 v8bf;
typedef __attribute__((ext_vector_type(4)))  __bf16 v4bf;
typedef __attribute__((ext_vector_type(8)))  float  v8f;

static constexpr int cB = 4, cN = 5000, cK = 4, cE = 80000, cD = 512, cL = 6, cLM = 3;
#define EPS_MSG 1e-7f
#define EPS_SM  1e-16f
#define EPS_LN  1e-5f
// 1/sqrt(1 + EPS_BN), EPS_BN = 1e-5
#define BN_INV  0.9999950000374997f

// ---- async global->LDS copy (CDNA5 GLOBAL_LOAD_ASYNC_TO_LDS_B128, ASYNCcnt)
#if __has_builtin(__builtin_amdgcn_global_load_async_to_lds_b128) && \
    __has_builtin(__builtin_amdgcn_s_wait_asynccnt)
#define HAVE_ASYNC_LDS 1
typedef int v4i_vs __attribute__((vector_size(4 * sizeof(int))));
typedef __attribute__((address_space(1))) v4i_vs as1_v4i;
typedef __attribute__((address_space(3))) v4i_vs as3_v4i;
__device__ __forceinline__ void async_cp16(const void* g, void* l) {
  __builtin_amdgcn_global_load_async_to_lds_b128(
      (as1_v4i*)(uintptr_t)g, (as3_v4i*)(uintptr_t)l, 0, 0);
}
#endif

// ---------------------------------------------------------------- utilities
__global__ void zero4_kernel(float4* __restrict__ p, int n4) {
  int i = blockIdx.x * blockDim.x + threadIdx.x;
  if (i < n4) p[i] = make_float4(0.f, 0.f, 0.f, 0.f);
}

// fp32 W[l][k][n] -> bf16 Wt[l][n][k]  (N-major so GEMM B-tiles stage as
// contiguous vector copies and fragments are contiguous ds_load_b128)
__global__ void cvt_wt_kernel(const float* __restrict__ W, __bf16* __restrict__ Wt,
                              int Kdim, int Nout, long total) {
  long i = (long)blockIdx.x * blockDim.x + threadIdx.x;
  if (i >= total) return;
  int  k = (int)(i % Kdim);
  long t = i / Kdim;
  int  n = (int)(t % Nout);
  long l = t / Nout;
  Wt[i] = (__bf16)W[(l * Kdim + k) * (long)Nout + n];
}

// ------------------------------------------------------ embedding gather-sum
__global__ void embed_kernel(const int* __restrict__ x_idx, const float* __restrict__ emb,
                             float* __restrict__ x) {
  const int D4 = cD / 4;
  int tid = blockIdx.x * blockDim.x + threadIdx.x;
  int total = cB * cN * D4;
  if (tid >= total) return;
  int d4  = tid % D4;
  int row = tid / D4;                       // b*N + n
  const int* ip = x_idx + (long)row * cK;
  float a = 0.f, b = 0.f, c = 0.f, d = 0.f;
#pragma unroll
  for (int k = 0; k < cK; ++k) {
    const float4 e = *(const float4*)(emb + (long)ip[k] * cD + d4 * 4);
    a += e.x; b += e.y; c += e.z; d += e.w;
  }
  *(float4*)(x + (long)row * cD + d4 * 4) = make_float4(a, b, c, d);
}

// --------------------------------------------------------- edge pass 1: max
// msg(b,e,d) = relu(x[b,src(e),d] + w(e)) + EPS_MSG  (> 0, so float max ==
// unsigned-int max on the bit pattern)
__global__ void edge_max_kernel(const float* __restrict__ x, const int* __restrict__ ei,
                                const float* __restrict__ ew, float* __restrict__ mx) {
  const int D4 = cD / 4;
  int tid = blockIdx.x * blockDim.x + threadIdx.x;
  int total = cB * cE * D4;
  if (tid >= total) return;
  int d4 = tid % D4;
  int t  = tid / D4;
  int e  = t % cE;
  int b  = t / cE;
  int s  = ei[e], dd = ei[cE + e];
  float w = ew[e];
  const float4 xv = *(const float4*)(x + ((long)b * cN + s) * cD + d4 * 4);
  unsigned* mp = (unsigned*)(mx + ((long)b * cN + dd) * cD + d4 * 4);
  atomicMax(mp + 0, __float_as_uint(fmaxf(xv.x + w, 0.f) + EPS_MSG));
  atomicMax(mp + 1, __float_as_uint(fmaxf(xv.y + w, 0.f) + EPS_MSG));
  atomicMax(mp + 2, __float_as_uint(fmaxf(xv.z + w, 0.f) + EPS_MSG));
  atomicMax(mp + 3, __float_as_uint(fmaxf(xv.w + w, 0.f) + EPS_MSG));
}

// ------------------------------------------- edge pass 2: exp-sum + weighted
__global__ void edge_sum_kernel(const float* __restrict__ x, const int* __restrict__ ei,
                                const float* __restrict__ ew, const float* __restrict__ mx,
                                float* __restrict__ S0, float* __restrict__ S1) {
  const int D4 = cD / 4;
  int tid = blockIdx.x * blockDim.x + threadIdx.x;
  int total = cB * cE * D4;
  if (tid >= total) return;
  int d4 = tid % D4;
  int t  = tid / D4;
  int e  = t % cE;
  int b  = t / cE;
  int s  = ei[e], dd = ei[cE + e];
  float w = ew[e];
  const float4 xv = *(const float4*)(x + ((long)b * cN + s) * cD + d4 * 4);
  long  doff = ((long)b * cN + dd) * cD + d4 * 4;
  const float4 mv = *(const float4*)(mx + doff);
  float* s0 = S0 + doff;
  float* s1 = S1 + doff;
  float m, ex;
  m = fmaxf(xv.x + w, 0.f) + EPS_MSG; ex = __expf(m - mv.x);
  atomicAdd(s0 + 0, ex); atomicAdd(s1 + 0, m * ex);
  m = fmaxf(xv.y + w, 0.f) + EPS_MSG; ex = __expf(m - mv.y);
  atomicAdd(s0 + 1, ex); atomicAdd(s1 + 1, m * ex);
  m = fmaxf(xv.z + w, 0.f) + EPS_MSG; ex = __expf(m - mv.z);
  atomicAdd(s0 + 2, ex); atomicAdd(s1 + 2, m * ex);
  m = fmaxf(xv.w + w, 0.f) + EPS_MSG; ex = __expf(m - mv.w);
  atomicAdd(s0 + 3, ex); atomicAdd(s1 + 3, m * ex);
}

// ------------------------------- combine: out = agg + x_in   (bf16 output)
__global__ void combine_kernel(const float* __restrict__ S0, const float* __restrict__ S1,
                               const float* __restrict__ xin, __bf16* __restrict__ xout,
                               int n4) {
  int i = blockIdx.x * blockDim.x + threadIdx.x;
  if (i >= n4) return;
  float4 s0 = ((const float4*)S0)[i];
  float4 s1 = ((const float4*)S1)[i];
  float4 xi = ((const float4*)xin)[i];
  v4bf p;
  p[0] = (__bf16)(s1.x / (s0.x + EPS_SM) + xi.x);
  p[1] = (__bf16)(s1.y / (s0.y + EPS_SM) + xi.y);
  p[2] = (__bf16)(s1.z / (s0.z + EPS_SM) + xi.z);
  p[3] = (__bf16)(s1.w / (s0.w + EPS_SM) + xi.w);
  *(v4bf*)(xout + (long)i * 4) = p;
}

// --------------------------------------------------------------- WMMA GEMM
// C(M x Nout) = A(M x Kdim bf16) @ W(Kdim x Nout) with W given as Wt[n][k]
// bf16.  Block tile 64x64, K-step 64; 8 waves; each wave owns a 16x32 output
// (2 accumulators), 4 v_wmma per staging round.  Double-buffered LDS with
// async global->LDS copies (ASYNCcnt): step i+1's copies overlap step i's
// wmmas; one barrier per K-step.
// epi: 0 = (v+b)*BN_INV*g+be, relu   1 = relu(v+b)   2 = gelu_exact(v+b)
// outBF: 1 -> store bf16, 0 -> store fp32
__global__ __launch_bounds__(256)
void gemm_bf16_kernel(const __bf16* __restrict__ A, const __bf16* __restrict__ Wt,
                      void* __restrict__ Cout, int M, int Kdim, int Nout,
                      int epi, int outBF, const float* __restrict__ bias,
                      const float* __restrict__ gamma, const float* __restrict__ beta) {
#ifdef HAVE_ASYNC_LDS
  __shared__ __align__(16) __bf16 lA[2][64][64];   // [buf][row][k]
  __shared__ __align__(16) __bf16 lBt[2][64][64];  // [buf][n][k]
#else
  __shared__ __align__(16) __bf16 lA[1][64][64];
  __shared__ __align__(16) __bf16 lBt[1][64][64];
#endif

  const int tid  = threadIdx.x;
  const int wave = tid >> 5;
  const int lane = tid & 31;
  const int wy   = wave >> 1;            // 0..3 : 16-row subtile
  const int wx   = wave & 1;             // 0..1 : 32-col subtile
  const int rowBase = blockIdx.x * 64;
  const int colBase = blockIdx.y * 64;
  const int half = lane >> 4;            // 0/1
  const int mn   = lane & 15;

  // staging coordinates: two 16B chunks per thread per tile
  const int c0 = tid,        r0 = c0 >> 3, k0 = (c0 & 7) * 8;
  const int c1 = tid + 256,  r1 = c1 >> 3, k1 = (c1 & 7) * 8;
  const __bf16* aBase0 = A  + (long)(rowBase + r0) * Kdim + k0;
  const __bf16* aBase1 = A  + (long)(rowBase + r1) * Kdim + k1;
  const __bf16* bBase0 = Wt + (long)(colBase + r0) * Kdim + k0;
  const __bf16* bBase1 = Wt + (long)(colBase + r1) * Kdim + k1;

  v8f acc0 = {};
  v8f acc1 = {};
  const int iters = Kdim >> 6;

#ifdef HAVE_ASYNC_LDS
  auto stage = [&](int buf, int kk) {
    async_cp16(aBase0 + kk, &lA[buf][r0][k0]);
    async_cp16(aBase1 + kk, &lA[buf][r1][k1]);
    async_cp16(bBase0 + kk, &lBt[buf][r0][k0]);
    async_cp16(bBase1 + kk, &lBt[buf][r1][k1]);
  };
  stage(0, 0);
#endif

  for (int i = 0; i < iters; ++i) {
#ifdef HAVE_ASYNC_LDS
    const int cur = i & 1;
    __builtin_amdgcn_s_wait_asynccnt(0);
    __syncthreads();                      // cur ready; nxt's readers all done
    if (i + 1 < iters) stage(cur ^ 1, (i + 1) << 6);
#else
    const int cur = 0;
    const int kk = i << 6;
    v8bf ra0 = *(const v8bf*)(aBase0 + kk);
    v8bf ra1 = *(const v8bf*)(aBase1 + kk);
    v8bf rb0 = *(const v8bf*)(bBase0 + kk);
    v8bf rb1 = *(const v8bf*)(bBase1 + kk);
    *(v8bf*)&lA[0][r0][k0]  = ra0;
    *(v8bf*)&lA[0][r1][k1]  = ra1;
    *(v8bf*)&lBt[0][r0][k0] = rb0;
    *(v8bf*)&lBt[0][r1][k1] = rb1;
    __syncthreads();
#endif

    const __bf16* aRow  = &lA[cur][16 * wy + mn][0];
    const __bf16* bRow0 = &lBt[cur][32 * wx + mn][0];
    const __bf16* bRow1 = &lBt[cur][32 * wx + 16 + mn][0];

    // gather all fragments first, then issue 4 back-to-back wmmas
    v16bf av[2], bv0[2], bv1[2];
#pragma unroll
    for (int kt = 0; kt < 2; ++kt) {
      // A frag: k = 16*half + t within k-tile (contiguous 16 values)
      v8bf a0 = *(const v8bf*)(aRow + kt * 32 + 16 * half);
      v8bf a1 = *(const v8bf*)(aRow + kt * 32 + 16 * half + 8);
      av[kt] = __builtin_shufflevector(a0, a1, 0,1,2,3,4,5,6,7,8,9,10,11,12,13,14,15);
      // B frag: k = 16*half + t within k-tile (contiguous 16 values)
      v8bf b0 = *(const v8bf*)(bRow0 + kt * 32 + 16 * half);
      v8bf b1 = *(const v8bf*)(bRow0 + kt * 32 + 16 * half + 8);
      bv0[kt] = __builtin_shufflevector(b0, b1, 0,1,2,3,4,5,6,7,8,9,10,11,12,13,14,15);
      v8bf b2 = *(const v8bf*)(bRow1 + kt * 32 + 16 * half);
      v8bf b3 = *(const v8bf*)(bRow1 + kt * 32 + 16 * half + 8);
      bv1[kt] = __builtin_shufflevector(b2, b3, 0,1,2,3,4,5,6,7,8,9,10,11,12,13,14,15);
    }
#pragma unroll
    for (int kt = 0; kt < 2; ++kt) {
      acc0 = __builtin_amdgcn_wmma_f32_16x16x32_bf16(false, av[kt], false, bv0[kt],
                                                     (short)0, acc0, false, false);
      acc1 = __builtin_amdgcn_wmma_f32_16x16x32_bf16(false, av[kt], false, bv1[kt],
                                                     (short)0, acc1, false, false);
    }
#ifndef HAVE_ASYNC_LDS
    __syncthreads();
#endif
  }

  // ---- epilogue: VGPR r -> m = r + 8*half, n = lane&15
#pragma unroll
  for (int nt = 0; nt < 2; ++nt) {
    const int col = colBase + 32 * wx + 16 * nt + mn;
    const float bs = bias[col];
    float gm = 0.f, bt = 0.f;
    if (epi == 0) { gm = gamma[col]; bt = beta[col]; }
#pragma unroll
    for (int r = 0; r < 8; ++r) {
      int grow = rowBase + 16 * wy + r + 8 * half;
      if (grow < M) {
        float v = (nt ? acc1[r] : acc0[r]) + bs;
        if (epi == 0) {
          v = fmaxf(v * BN_INV * gm + bt, 0.f);
        } else if (epi == 1) {
          v = fmaxf(v, 0.f);
        } else {
          v = 0.5f * v * (1.f + erff(v * 0.70710678118654752f));
        }
        long idx = (long)grow * Nout + col;
        if (outBF) ((__bf16*)Cout)[idx] = (__bf16)v;
        else       ((float*)Cout)[idx]  = v;
      }
    }
  }
}

// ------------------------------------------------- LayerNorm (bf16 output)
__global__ void layernorm_kernel(const float* __restrict__ x, __bf16* __restrict__ y,
                                 const float* __restrict__ g, const float* __restrict__ b,
                                 int rows) {
  int warp = threadIdx.x >> 5, lane = threadIdx.x & 31;
  int row = blockIdx.x * 8 + warp;
  if (row >= rows) return;
  const float* xr = x + (long)row * cD;
  float v[16];
  float s = 0.f;
#pragma unroll
  for (int i = 0; i < 16; ++i) { v[i] = xr[lane + 32 * i]; s += v[i]; }
#pragma unroll
  for (int o = 16; o > 0; o >>= 1) s += __shfl_xor(s, o, 32);
  float mu = s * (1.f / cD);
  float q = 0.f;
#pragma unroll
  for (int i = 0; i < 16; ++i) { float d = v[i] - mu; q += d * d; }
#pragma unroll
  for (int o = 16; o > 0; o >>= 1) q += __shfl_xor(q, o, 32);
  float inv = rsqrtf(q * (1.f / cD) + EPS_LN);
  __bf16* yr = y + (long)row * cD;
#pragma unroll
  for (int i = 0; i < 16; ++i) {
    int d = lane + 32 * i;
    yr[d] = (__bf16)((v[i] - mu) * inv * g[d] + b[d]);
  }
}

// --------------------------------------------------------- final projection
__global__ void out_init_kernel(float* __restrict__ out, const float* __restrict__ bout) {
  if (threadIdx.x < cB && blockIdx.x == 0) out[threadIdx.x] = bout[0];
}

__global__ void out_reduce_kernel(const float* __restrict__ h, const float* __restrict__ Wout,
                                  float* __restrict__ out) {
  int warp = threadIdx.x >> 5, lane = threadIdx.x & 31;
  long row = (long)blockIdx.x * 8 + warp;     // row in [0, B*N)
  if (row >= (long)cB * cN) return;
  const float* hr = h + row * cD;
  float s = 0.f;
#pragma unroll
  for (int i = 0; i < 16; ++i) {
    int d = lane + 32 * i;
    s += hr[d] * Wout[d];
  }
#pragma unroll
  for (int o = 16; o > 0; o >>= 1) s += __shfl_xor(s, o, 32);
  if (lane == 0) atomicAdd(&out[row / cN], s * (1.f / cN));
}

// ------------------------------------------------------------------ driver
extern "C" void kernel_launch(void* const* d_in, const int* in_sizes, int n_in,
                              void* d_out, int out_size, void* d_ws, size_t ws_size,
                              hipStream_t stream) {
  (void)in_sizes; (void)n_in; (void)out_size; (void)ws_size;

  const int*   x_idx = (const int*)  d_in[0];
  const int*   ei    = (const int*)  d_in[1];
  const float* ew    = (const float*)d_in[2];
  const float* emb   = (const float*)d_in[3];
  const float* W1    = (const float*)d_in[4];
  const float* b1    = (const float*)d_in[5];
  const float* g1    = (const float*)d_in[6];
  const float* be1   = (const float*)d_in[7];
  const float* W2    = (const float*)d_in[8];
  const float* b2    = (const float*)d_in[9];
  const float* ln_g  = (const float*)d_in[10];
  const float* ln_b  = (const float*)d_in[11];
  const float* Wm    = (const float*)d_in[12];
  const float* bm    = (const float*)d_in[13];
  const float* Wout  = (const float*)d_in[14];
  const float* bout  = (const float*)d_in[15];
  float* out = (float*)d_out;

  const size_t BND = (size_t)cB * cN * cD;       // 10,240,000
  char* ws = (char*)d_ws;
  float*  xA   = (float*)ws;  ws += BND * 4;     // fp32 activations (edge phase)
  float*  mx   = (float*)ws;  ws += BND * 4;     // mx,S0,S1 contiguous
  float*  S0   = (float*)ws;  ws += BND * 4;
  float*  S1   = (float*)ws;  ws += BND * 4;
  __bf16* xBbf = (__bf16*)ws; ws += BND * 2;     // combine out / gemm1 in / mlp ping
  __bf16* hbf  = (__bf16*)ws; ws += 2 * BND * 2; // gemm1 out / gemm2 in; LN+MLP aliases
  __bf16* W1t  = (__bf16*)ws; ws += (size_t)cL * cD * 2 * cD * 2;   // [l][n=1024][k=512]
  __bf16* W2t  = (__bf16*)ws; ws += (size_t)cL * 2 * cD * cD * 2;   // [l][n=512][k=1024]
  __bf16* Wmt  = (__bf16*)ws; ws += (size_t)cLM * cD * cD * 2;      // [l][n=512][k=512]
  __bf16* lnbf = hbf;            // LN out / mlp0 in
  __bf16* m0bf = xBbf;           // mlp0 out / mlp1 in
  __bf16* m1bf = hbf + BND;      // mlp1 out / mlp2 in

  const int M = cB * cN;                         // 20000 GEMM rows
  const int TB = 256;
  const int gemmGX = (M + 63) / 64;              // 313

  // weights -> transposed bf16 (once per launch; deterministic)
  {
    long n1 = (long)cL * cD * 2 * cD;
    cvt_wt_kernel<<<(int)((n1 + TB - 1) / TB), TB, 0, stream>>>(W1, W1t, cD, 2 * cD, n1);
    long n2 = (long)cL * 2 * cD * cD;
    cvt_wt_kernel<<<(int)((n2 + TB - 1) / TB), TB, 0, stream>>>(W2, W2t, 2 * cD, cD, n2);
    long n3 = (long)cLM * cD * cD;
    cvt_wt_kernel<<<(int)((n3 + TB - 1) / TB), TB, 0, stream>>>(Wm, Wmt, cD, cD, n3);
  }

  // x = node_emb[x_idx].sum(axis=2)
  {
    int total = cB * cN * (cD / 4);
    embed_kernel<<<(total + TB - 1) / TB, TB, 0, stream>>>(x_idx, emb, xA);
  }

  const int edgeTotal = cB * cE * (cD / 4);      // 40,960,000
  const int edgeBlocks = (edgeTotal + TB - 1) / TB;
  const int n4 = (int)(BND / 4);

  for (int l = 0; l < cL; ++l) {
    // zero mx,S0,S1 (contiguous 3*BND floats)
    int z4 = (int)(3 * BND / 4);
    zero4_kernel<<<(z4 + TB - 1) / TB, TB, 0, stream>>>((float4*)mx, z4);

    edge_max_kernel<<<edgeBlocks, TB, 0, stream>>>(xA, ei, ew, mx);
    edge_sum_kernel<<<edgeBlocks, TB, 0, stream>>>(xA, ei, ew, mx, S0, S1);
    combine_kernel<<<(n4 + TB - 1) / TB, TB, 0, stream>>>(S0, S1, xA, xBbf, n4);

    // h = relu(bn(xB @ W1 + b1))        (20000 x 512) @ (512 x 1024), bf16 out
    gemm_bf16_kernel<<<dim3(gemmGX, (2 * cD) / 64), TB, 0, stream>>>(
        xBbf, W1t + (size_t)l * cD * 2 * cD, hbf, M, cD, 2 * cD,
        0, 1, b1 + (size_t)l * 2 * cD, g1 + (size_t)l * 2 * cD, be1 + (size_t)l * 2 * cD);

    // x = relu(h @ W2 + b2)             (20000 x 1024) @ (1024 x 512), fp32 out
    gemm_bf16_kernel<<<dim3(gemmGX, cD / 64), TB, 0, stream>>>(
        hbf, W2t + (size_t)l * 2 * cD * cD, xA, M, 2 * cD, cD,
        1, 0, b2 + (size_t)l * cD, nullptr, nullptr);
  }

  // LayerNorm: xA -> lnbf (bf16)
  layernorm_kernel<<<(M + 7) / 8, TB, 0, stream>>>(xA, lnbf, ln_g, ln_b, M);

  // MLP: gelu GEMMs: lnbf -> m0bf -> m1bf -> xA(fp32)
  gemm_bf16_kernel<<<dim3(gemmGX, cD / 64), TB, 0, stream>>>(
      lnbf, Wmt + (size_t)0 * cD * cD, m0bf, M, cD, cD, 2, 1, bm + 0 * cD, nullptr, nullptr);
  gemm_bf16_kernel<<<dim3(gemmGX, cD / 64), TB, 0, stream>>>(
      m0bf, Wmt + (size_t)1 * cD * cD, m1bf, M, cD, cD, 2, 1, bm + 1 * cD, nullptr, nullptr);
  gemm_bf16_kernel<<<dim3(gemmGX, cD / 64), TB, 0, stream>>>(
      m1bf, Wmt + (size_t)2 * cD * cD, xA, M, cD, cD, 2, 0, bm + 2 * cD, nullptr, nullptr);

  // out[b] = bout + mean_n( h[b,n,:] . Wout )
  out_init_kernel<<<1, 32, 0, stream>>>(out, bout);
  out_reduce_kernel<<<(M + 7) / 8, TB, 0, stream>>>(xA, Wout, out);
}
